// VGCNEncoder_38654705664008
// MI455X (gfx1250) — compile-verified, matrix operand order checked by reading, not run on hardware
//
#include <hip/hip_runtime.h>
#include <hip/hip_bf16.h>

#define NN 50000
#define NE 800000
#define INDIM 128
#define HID 64

typedef __attribute__((ext_vector_type(16))) _Float16 v16h;
typedef __attribute__((ext_vector_type(8)))  float    v8f;

// ---------------- degree / normalization ----------------
__global__ void deg_init_kernel(float* deg, int n) {
    int i = blockIdx.x * blockDim.x + threadIdx.x;
    if (i < n) deg[i] = 1.0f;
}

__global__ void deg_accum_kernel(const int* __restrict__ dst, float* deg, int e) {
    int i = blockIdx.x * blockDim.x + threadIdx.x;
    if (i < e) atomicAdd(&deg[dst[i]], 1.0f);
}

__global__ void dis_kernel(float* deg, int n) {
    int i = blockIdx.x * blockDim.x + threadIdx.x;
    if (i < n) deg[i] = rsqrtf(deg[i]);
}

// ---------------- WMMA GEMM:  C[M x 64] = A[M x K] @ W[K x 64] ----------------
// 256 threads = 8 waves (wave32). Each wave owns 16 rows x 64 cols (4 n-tiles).
// K-loop of v_wmma_f32_16x16x32_f16, f32 accumulate. W staged to LDS as f16,
// n-major with pad so each lane's 8-half K-run is contiguous.
template<int K>
__global__ __launch_bounds__(256) void gemm_wmma_kernel(
    const float* __restrict__ A, const float* __restrict__ W,
    float* __restrict__ C, int M)
{
    constexpr int KP = K + 8;
    __shared__ _Float16 Wl[64 * KP];

    // stage W (row-major K x 64, f32) -> Wl[n*KP + k] as f16
    for (int idx = threadIdx.x; idx < K * 64; idx += 256) {
        int k = idx >> 6;
        int n = idx & 63;
        Wl[n * KP + k] = (_Float16)W[idx];
    }
    __syncthreads();

    const int wave     = threadIdx.x >> 5;
    const int lane     = threadIdx.x & 31;
    const int half_sel = lane >> 4;        // 0: lanes 0-15, 1: lanes 16-31
    const int lrow     = lane & 15;
    const int klo      = half_sel * 8;

    int row  = blockIdx.x * 128 + wave * 16 + lrow;
    int rowc = row < M ? row : (M - 1);    // clamp: keep EXEC all-1s for WMMA
    const float* __restrict__ Arow = A + (size_t)rowc * K;

    v8f acc[4];
    acc[0] = (v8f){}; acc[1] = (v8f){}; acc[2] = (v8f){}; acc[3] = (v8f){};

    #pragma unroll
    for (int kb = 0; kb < K; kb += 32) {
        // A fragment per ISA 16-bit A 16x32 layout
        v16h a;
        {
            const float4* p = (const float4*)(Arow + kb + klo);
            float4 q0 = p[0];
            float4 q1 = p[1];
            const float4* r = (const float4*)(Arow + kb + 16 + klo);
            float4 q2 = r[0];
            float4 q3 = r[1];
            a[0]  = (_Float16)q0.x; a[1]  = (_Float16)q0.y;
            a[2]  = (_Float16)q0.z; a[3]  = (_Float16)q0.w;
            a[4]  = (_Float16)q1.x; a[5]  = (_Float16)q1.y;
            a[6]  = (_Float16)q1.z; a[7]  = (_Float16)q1.w;
            a[8]  = (_Float16)q2.x; a[9]  = (_Float16)q2.y;
            a[10] = (_Float16)q2.z; a[11] = (_Float16)q2.w;
            a[12] = (_Float16)q3.x; a[13] = (_Float16)q3.y;
            a[14] = (_Float16)q3.z; a[15] = (_Float16)q3.w;
        }
        #pragma unroll
        for (int nt = 0; nt < 4; ++nt) {
            const int ncol = nt * 16 + lrow;
            const _Float16* __restrict__ bp = &Wl[ncol * KP + kb + klo];
            v16h b;
            #pragma unroll
            for (int hh = 0; hh < 8; ++hh) {
                b[hh]     = bp[hh];        // K = kb+klo   .. +7
                b[hh + 8] = bp[16 + hh];   // K = kb+16+klo.. +7
            }
            acc[nt] = __builtin_amdgcn_wmma_f32_16x16x32_f16(
                false, a, false, b, (short)0, acc[nt], false, false);
        }
    }

    // D layout: VGPR g -> row (g + half_sel*8), col = lane&15 within n-tile
    const int rbase = blockIdx.x * 128 + wave * 16 + half_sel * 8;
    #pragma unroll
    for (int g = 0; g < 8; ++g) {
        int r = rbase + g;
        if (r < M) {
            float* crow = C + (size_t)r * 64 + lrow;
            crow[0]  = acc[0][g];
            crow[16] = acc[1][g];
            crow[32] = acc[2][g];
            crow[48] = acc[3][g];
        }
    }
}

// ---------------- elementwise / scatter ----------------
// out[i] = t[i] * dis[row]^2 + bias[col]   (self-loop term + bias)
__global__ void self_bias_kernel(const float* __restrict__ t,
                                 const float* __restrict__ dis,
                                 const float* __restrict__ bias,
                                 float* __restrict__ out, int total)
{
    int i = blockIdx.x * blockDim.x + threadIdx.x;
    if (i < total) {
        int row = i >> 6;
        int c   = i & 63;
        float d = dis[row];
        out[i] = t[i] * d * d + bias[c];
    }
}

// 16 threads per edge, float4 per thread: gather t[src] * dis[src]*dis[dst],
// scatter-add into out[dst]. Atomics land in L2 (t fits in 192MB L2).
__global__ void edge_scatter_kernel(const int* __restrict__ src,
                                    const int* __restrict__ dst,
                                    const float* __restrict__ dis,
                                    const float* __restrict__ t,
                                    float* out, int e)
{
    int tid = blockIdx.x * blockDim.x + threadIdx.x;
    int ed  = tid >> 4;
    if (ed < e) {
        int c4 = (tid & 15) * 4;
        int s = src[ed];
        int d = dst[ed];
        float w = dis[s] * dis[d];
        float4 v = *(const float4*)(t + (size_t)s * 64 + c4);
        float* o = out + (size_t)d * 64 + c4;
        atomicAdd(o + 0, v.x * w);
        atomicAdd(o + 1, v.y * w);
        atomicAdd(o + 2, v.z * w);
        atomicAdd(o + 3, v.w * w);
    }
}

__global__ void relu_kernel(float* h, int total) {
    int i = blockIdx.x * blockDim.x + threadIdx.x;
    if (i < total) h[i] = fmaxf(h[i], 0.0f);
}

// ---------------- driver ----------------
extern "C" void kernel_launch(void* const* d_in, const int* in_sizes, int n_in,
                              void* d_out, int out_size, void* d_ws, size_t ws_size,
                              hipStream_t stream)
{
    const float* x   = (const float*)d_in[0];
    const int*   ei  = (const int*)d_in[1];
    const float* W1  = (const float*)d_in[2];
    const float* b1  = (const float*)d_in[3];
    const float* Wmu = (const float*)d_in[4];
    const float* bmu = (const float*)d_in[5];
    const float* Wlv = (const float*)d_in[6];
    const float* blv = (const float*)d_in[7];

    const int* src = ei;            // edge_index[0]
    const int* dst = ei + NE;       // edge_index[1]

    float* out_mu = (float*)d_out;
    float* out_lv = out_mu + (size_t)NN * HID;

    float* ws  = (float*)d_ws;
    float* dis = ws;                        // N floats (deg -> dis in place)
    float* t   = ws + 50048;                // N*64: GEMM output (reused 3x)
    float* h   = t + (size_t)NN * HID;      // N*64: hidden after conv1+relu

    const int total = NN * HID;
    dim3 blk(256);

    // normalization
    deg_init_kernel <<<(NN + 255) / 256, blk, 0, stream>>>(dis, NN);
    deg_accum_kernel<<<(NE + 255) / 256, blk, 0, stream>>>(dst, dis, NE);
    dis_kernel      <<<(NN + 255) / 256, blk, 0, stream>>>(dis, NN);

    const int gemm_grid    = (NN + 127) / 128;
    const int ew_grid      = (total + 255) / 256;
    const int scatter_grid = ((NE * 16) + 255) / 256;

    // conv1: h = relu(agg(x @ W1) + self + b1)
    gemm_wmma_kernel<INDIM><<<gemm_grid, blk, 0, stream>>>(x, W1, t, NN);
    self_bias_kernel   <<<ew_grid,      blk, 0, stream>>>(t, dis, b1, h, total);
    edge_scatter_kernel<<<scatter_grid, blk, 0, stream>>>(src, dst, dis, t, h, NE);
    relu_kernel        <<<ew_grid,      blk, 0, stream>>>(h, total);

    // conv mu -> d_out[0 : N*64]
    gemm_wmma_kernel<HID><<<gemm_grid, blk, 0, stream>>>(h, Wmu, t, NN);
    self_bias_kernel   <<<ew_grid,      blk, 0, stream>>>(t, dis, bmu, out_mu, total);
    edge_scatter_kernel<<<scatter_grid, blk, 0, stream>>>(src, dst, dis, t, out_mu, NE);

    // conv logvar -> d_out[N*64 : 2*N*64]
    gemm_wmma_kernel<HID><<<gemm_grid, blk, 0, stream>>>(h, Wlv, t, NN);
    self_bias_kernel   <<<ew_grid,      blk, 0, stream>>>(t, dis, blv, out_lv, total);
    edge_scatter_kernel<<<scatter_grid, blk, 0, stream>>>(src, dst, dis, t, out_lv, NE);
}
